// SparseMoE_37658273251435
// MI455X (gfx1250) — matrix operand order repaired and checked by reading, not compiled, not run on hardware
//
#include <hip/hip_runtime.h>

// Problem constants (match reference)
#define NTOK 8192      // B*T
#define CDIM 1024      // C
#define DFF  4096
#define NE   8         // experts
#define CAP  NTOK      // per-expert list capacity (worst case)
#define MT   64        // tokens per expert block (M tile)

typedef __bf16 bf16;
typedef __attribute__((ext_vector_type(16))) bf16  v16bf;
typedef __attribute__((ext_vector_type(4)))  bf16  v4bf;
typedef __attribute__((ext_vector_type(8)))  float v8f;

union AFrag { v16bf v; uint4 u[2]; };

// LDS geometry (bf16 elements). Strides keep 16B alignment for ds_load_b128.
#define XT_STRIDE 1032          // 64 rows x 1024 (+8 pad); 2064B = 16*129
#define HT_STRIDE 264           // 64 rows x 256  (+8 pad); 528B  = 16*33
#define SMEM_BYTES ((MT * XT_STRIDE + MT * HT_STRIDE) * 2)   // 165,888 B

static __device__ __forceinline__ v8f zero8() {
    v8f z;
#pragma unroll
    for (int i = 0; i < 8; ++i) z[i] = 0.0f;
    return z;
}

static __device__ __forceinline__ v8f wmma_bf16(const AFrag& a, const AFrag& b, v8f c) {
    return __builtin_amdgcn_wmma_f32_16x16x32_bf16(false, a.v, false, b.v,
                                                   (short)0, c, false, false);
}

// ---------------------------------------------------------------------------
// K0: fp32 -> bf16 for x; zero output and expert counters.
// ---------------------------------------------------------------------------
__global__ __launch_bounds__(256) void k_prep(
    const float* __restrict__ x, bf16* __restrict__ xb,
    float* __restrict__ out, int* __restrict__ cnt)
{
    size_t tid = (size_t)blockIdx.x * blockDim.x + threadIdx.x;
    size_t stride = (size_t)gridDim.x * blockDim.x;
    const size_t NX4 = (size_t)NTOK * CDIM / 4;         // 2M float4
    for (size_t j = tid; j < NX4; j += stride) {
        float4 a = ((const float4*)x)[j];
        v4bf p;
        p[0] = (bf16)a.x; p[1] = (bf16)a.y; p[2] = (bf16)a.z; p[3] = (bf16)a.w;
        ((v4bf*)xb)[j] = p;
        float4 z; z.x = z.y = z.z = z.w = 0.0f;
        ((float4*)out)[j] = z;                           // out has NTOK*C floats
    }
    if (tid < NE) cnt[tid] = 0;
}

// ---------------------------------------------------------------------------
// K1: fp32 -> bf16 weight conversion + swizzle into B-fragment order.
// Element (K-row r, col c) of a (R x L) matrix goes to
//   dst[ ((e*(R/32) + r/32)*L + c)*32 + (r%32) ]
// so each column's 32 K-values are contiguous (64B) per K-chunk.
// One block per 32x256 tile. y==0: w1 (C x DFF), y==1: w2 (DFF x C).
// ---------------------------------------------------------------------------
__global__ __launch_bounds__(256) void k_swz(
    const float* __restrict__ w1, const float* __restrict__ w2,
    bf16* __restrict__ w1s, bf16* __restrict__ w2s)
{
    const int tid = threadIdx.x;
    const int t = blockIdx.x;
    const int e = t >> 9, rem = t & 511;     // 512 tiles per expert per matrix
    const float* src; bf16* dst;
    int L, r32, cb;
    if (blockIdx.y == 0) {
        L = DFF; r32 = rem >> 4; cb = rem & 15;          // 32 K-chunks x 16 col-blocks
        src = w1 + (size_t)e * CDIM * DFF;
        dst = w1s + ((((size_t)e * (CDIM / 32) + r32) * DFF) << 5);
    } else {
        L = CDIM; r32 = rem >> 2; cb = rem & 3;          // 128 K-chunks x 4 col-blocks
        src = w2 + (size_t)e * DFF * CDIM;
        dst = w2s + ((((size_t)e * (DFF / 32) + r32) * CDIM) << 5);
    }
    const int c0 = cb * 256;
    const float* s = src + (size_t)(r32 * 32) * L + c0;
    // 32 rows x 256 cols; thread i: ci = i&31 (K row), f4 = i>>5 (col/4).
    // Writes: a wave (fixed f4, ci=0..31) stores 64B contiguous runs -> coalesced.
#pragma unroll
    for (int it = 0; it < 8; ++it) {
        int i = tid + it * 256;
        int ci = i & 31, f4 = i >> 5;
        float4 a = *(const float4*)(s + (size_t)ci * L + f4 * 4);
        bf16* d = dst + (((size_t)(c0 + f4 * 4)) << 5) + ci;
        d[0]  = (bf16)a.x;
        d[32] = (bf16)a.y;
        d[64] = (bf16)a.z;
        d[96] = (bf16)a.w;
    }
}

// ---------------------------------------------------------------------------
// K2: noisy top-2 router. One wave per token.
// ---------------------------------------------------------------------------
__global__ __launch_bounds__(256) void k_router(
    const float* __restrict__ x, const float* __restrict__ noise,
    const float* __restrict__ wg, const float* __restrict__ bg,
    const float* __restrict__ wn, const float* __restrict__ bn,
    int* __restrict__ cnt, int* __restrict__ tokIds, float* __restrict__ tokWs)
{
    const int wv = threadIdx.x >> 5, lane = threadIdx.x & 31;
    const int t = blockIdx.x * 8 + wv;
    const float* xr = x + (size_t)t * CDIM;

    float ag[NE], an[NE];
#pragma unroll
    for (int e = 0; e < NE; ++e) { ag[e] = 0.0f; an[e] = 0.0f; }

    for (int c = lane; c < CDIM; c += 32) {
        float xv = xr[c];
        const float4* g  = (const float4*)(wg + (size_t)c * NE);
        const float4* nz = (const float4*)(wn + (size_t)c * NE);
        float4 g0 = g[0], g1 = g[1], n0 = nz[0], n1 = nz[1];
        ag[0] += xv * g0.x; ag[1] += xv * g0.y; ag[2] += xv * g0.z; ag[3] += xv * g0.w;
        ag[4] += xv * g1.x; ag[5] += xv * g1.y; ag[6] += xv * g1.z; ag[7] += xv * g1.w;
        an[0] += xv * n0.x; an[1] += xv * n0.y; an[2] += xv * n0.z; an[3] += xv * n0.w;
        an[4] += xv * n1.x; an[5] += xv * n1.y; an[6] += xv * n1.z; an[7] += xv * n1.w;
    }
#pragma unroll
    for (int e = 0; e < NE; ++e) {
        for (int off = 16; off > 0; off >>= 1) {
            ag[e] += __shfl_down(ag[e], off, 32);
            an[e] += __shfl_down(an[e], off, 32);
        }
    }
    if (lane == 0) {
        float logit[NE];
#pragma unroll
        for (int e = 0; e < NE; ++e) {
            float cl = ag[e] + bg[e];
            float nl = an[e] + bn[e];
            float sp = (nl > 20.0f) ? nl : log1pf(expf(nl));   // softplus
            logit[e] = cl + noise[(size_t)t * NE + e] * sp;
        }
        int i0 = 0;
#pragma unroll
        for (int e = 1; e < NE; ++e) if (logit[e] > logit[i0]) i0 = e;
        int i1 = (i0 == 0) ? 1 : 0;
#pragma unroll
        for (int e = 0; e < NE; ++e) if (e != i0 && logit[e] > logit[i1]) i1 = e;
        float m  = fmaxf(logit[i0], logit[i1]);
        float e0 = expf(logit[i0] - m), e1 = expf(logit[i1] - m);
        float inv = 1.0f / (e0 + e1);
        int p0 = atomicAdd(&cnt[i0], 1);
        tokIds[(size_t)i0 * CAP + p0] = t; tokWs[(size_t)i0 * CAP + p0] = e0 * inv;
        int p1 = atomicAdd(&cnt[i1], 1);
        tokIds[(size_t)i1 * CAP + p1] = t; tokWs[(size_t)i1 * CAP + p1] = e1 * inv;
    }
}

// ---------------------------------------------------------------------------
// K3: fused expert FFN, 64 gathered tokens per block, 16 waves, N=1024 in regs.
// Wave w owns f-slice [w*16] for GEMM1 and col-slice [w*64] for GEMM2, so each
// pre-swizzled B fragment is loaded from L2 exactly once per block. A-fragment
// re-reads land in LDS. Weight L2 traffic: (16384/64) * 16MB = 4 GB total.
// ---------------------------------------------------------------------------
__global__ __launch_bounds__(512) void k_expert(
    const bf16* __restrict__ w1s, const bf16* __restrict__ w2s,
    const bf16* __restrict__ xb,
    const float* __restrict__ b1, const float* __restrict__ b2,
    const int* __restrict__ cnt, const int* __restrict__ tokIds,
    const float* __restrict__ tokWs, float* __restrict__ out)
{
    extern __shared__ bf16 smem[];
    bf16* xt = smem;                  // 64 x XT_STRIDE  (gathered x tile)
    bf16* ht = xt + MT * XT_STRIDE;   // 64 x HT_STRIDE  (h chunk)
    __shared__ int   toks[MT];
    __shared__ float wts[MT];

    const int e = blockIdx.y;
    const int n = cnt[e];
    const int tile = blockIdx.x;
    if (tile * MT >= n) return;

    const int tid  = threadIdx.x;     // 0..511
    const int wv   = tid >> 5;        // 0..15
    const int lane = tid & 31;
    const int nn   = lane & 15;       // N (or M) index within 16-wide tile
    const int half = lane >> 4;

    if (tid < MT) {
        int r = tile * MT + tid;
        bool ok = r < n;
        toks[tid] = ok ? tokIds[(size_t)e * CAP + r] : 0;
        wts[tid]  = ok ? tokWs[(size_t)e * CAP + r] : 0.0f;
    }
    __syncthreads();

    // Gather x tile (64 x 1024 bf16) into LDS with b128 moves; uniform trip.
#pragma unroll
    for (int it = 0; it < 16; ++it) {
        int i = tid + it * 512;
        int row = i >> 7, c16 = i & 127;
        *(uint4*)(xt + row * XT_STRIDE + c16 * 8) =
            *(const uint4*)(xb + (size_t)toks[row] * CDIM + c16 * 8);
    }
    __syncthreads();

    const bf16* w1e = w1s + (((size_t)e * (CDIM / 32)) * DFF << 5);
    const bf16* w2e = w2s + (((size_t)e * (DFF / 32)) * CDIM << 5);

    v8f yacc[16];                     // 4 M-subtiles x 4 N-tiles (cols wv*64..+64)
#pragma unroll
    for (int t = 0; t < 16; ++t) yacc[t] = zero8();

    for (int fc = 0; fc < DFF / 256; ++fc) {
        const int f0 = fc * 256;
        v8f hacc[4];                  // h rows (4 M-subtiles) x 16 f-cols (wv*16)
#pragma unroll
        for (int mi = 0; mi < 4; ++mi) hacc[mi] = zero8();

        // ---- GEMM1: h(64x256) = x(64x1024) @ w1[:, f0:f0+256], K = 1024
        {
            const bf16* bp0 = w1e + (((size_t)(f0 + wv * 16 + nn)) << 5) + half * 8;
#pragma unroll 2
            for (int kc32 = 0; kc32 < CDIM / 32; ++kc32) {
                AFrag b;              // B loaded ONCE per block per k-step
                const bf16* p = bp0 + (((size_t)kc32 * DFF) << 5);
                b.u[0] = *(const uint4*)(p);
                b.u[1] = *(const uint4*)(p + 16);
#pragma unroll
                for (int mi = 0; mi < 4; ++mi) {
                    AFrag a;
                    const bf16* ap = xt + (mi * 16 + nn) * XT_STRIDE + kc32 * 32 + half * 8;
                    a.u[0] = *(const uint4*)(ap);
                    a.u[1] = *(const uint4*)(ap + 16);
                    hacc[mi] = wmma_bf16(a, b, hacc[mi]);
                }
            }
        }

        __syncthreads();   // previous f-chunk's GEMM2 reads of ht are done
        // ReLU + bias, store h chunk (D layout: lane=N, vgpr j -> M=j+8*half)
        {
            float b1v = b1[(size_t)e * DFF + f0 + wv * 16 + nn];
#pragma unroll
            for (int mi = 0; mi < 4; ++mi) {
#pragma unroll
                for (int j = 0; j < 8; ++j) {
                    int row = mi * 16 + j + 8 * half;
                    float v = hacc[mi][j] + b1v; if (v < 0.0f) v = 0.0f;
                    ht[row * HT_STRIDE + wv * 16 + nn] = (bf16)v;
                }
            }
        }
        __syncthreads();   // h chunk visible to all waves

        // ---- GEMM2: y(64x1024) += h(64x256) @ w2[f0:f0+256, :], K = 256
        for (int ks32 = 0; ks32 < 8; ++ks32) {
            AFrag barr[4];            // 4 B tiles (cols wv*64..+64), each loaded once
            const bf16* bp = w2e +
                (((size_t)(f0 / 32 + ks32) * CDIM + wv * 64 + nn) << 5) + half * 8;
#pragma unroll
            for (int t = 0; t < 4; ++t) {
                const bf16* p = bp + ((t * 16) << 5);
                barr[t].u[0] = *(const uint4*)(p);
                barr[t].u[1] = *(const uint4*)(p + 16);
            }
#pragma unroll
            for (int mi = 0; mi < 4; ++mi) {
                AFrag a;
                const bf16* ap = ht + (mi * 16 + nn) * HT_STRIDE + ks32 * 32 + half * 8;
                a.u[0] = *(const uint4*)(ap);
                a.u[1] = *(const uint4*)(ap + 16);
#pragma unroll
                for (int t = 0; t < 4; ++t)
                    yacc[mi * 4 + t] = wmma_bf16(a, barr[t], yacc[mi * 4 + t]);
            }
        }
    }

    // Epilogue: out[token, c] += weight * (y + b2[c]); each token hit by 2 experts.
#pragma unroll
    for (int t = 0; t < 4; ++t) {
        int c = wv * 64 + t * 16 + nn;
        float b2v = b2[(size_t)e * CDIM + c];
#pragma unroll
        for (int mi = 0; mi < 4; ++mi) {
#pragma unroll
            for (int j = 0; j < 8; ++j) {
                int row = mi * 16 + j + 8 * half;
                float val = (yacc[mi * 4 + t][j] + b2v) * wts[row];
                atomicAdd(out + (size_t)toks[row] * CDIM + c, val);
            }
        }
    }
}

// ---------------------------------------------------------------------------
extern "C" void kernel_launch(void* const* d_in, const int* in_sizes, int n_in,
                              void* d_out, int out_size, void* d_ws, size_t ws_size,
                              hipStream_t stream) {
    (void)in_sizes; (void)n_in; (void)out_size; (void)ws_size;
    const float* x       = (const float*)d_in[0];
    const float* noise   = (const float*)d_in[1];
    const float* w_gate  = (const float*)d_in[2];
    const float* b_gate  = (const float*)d_in[3];
    const float* w_noise = (const float*)d_in[4];
    const float* b_noise = (const float*)d_in[5];
    const float* w1      = (const float*)d_in[6];
    const float* b1      = (const float*)d_in[7];
    const float* w2      = (const float*)d_in[8];
    const float* b2      = (const float*)d_in[9];
    float* out = (float*)d_out;

    char* ws = (char*)d_ws;
    size_t off = 0;
    bf16* w1s = (bf16*)(ws + off); off += (size_t)NE * CDIM * DFF * 2;   // 64 MB
    bf16* w2s = (bf16*)(ws + off); off += (size_t)NE * DFF * CDIM * 2;   // 64 MB
    bf16* xb  = (bf16*)(ws + off); off += (size_t)NTOK * CDIM * 2;       // 16 MB
    int*   cnt    = (int*)(ws + off);   off += 256;
    int*   tokIds = (int*)(ws + off);   off += (size_t)NE * CAP * 4;     // 256 KB
    float* tokWs  = (float*)(ws + off); off += (size_t)NE * CAP * 4;     // 256 KB

    hipLaunchKernelGGL(k_prep, dim3(2048), dim3(256), 0, stream, x, xb, out, cnt);
    hipLaunchKernelGGL(k_swz, dim3(4096, 2), dim3(256), 0, stream, w1, w2, w1s, w2s);
    hipLaunchKernelGGL(k_router, dim3(NTOK / 8), dim3(256), 0, stream,
                       x, noise, w_gate, b_gate, w_noise, b_noise,
                       cnt, tokIds, tokWs);
    hipFuncSetAttribute(reinterpret_cast<const void*>(k_expert),
                        hipFuncAttributeMaxDynamicSharedMemorySize, SMEM_BYTES);
    hipLaunchKernelGGL(k_expert, dim3(NTOK / MT, NE), dim3(512), SMEM_BYTES, stream,
                       w1s, w2s, xb, b1, b2, cnt, tokIds, tokWs, out);
}
